// Encoder_3135326126818
// MI455X (gfx1250) — matrix operand order repaired and checked by reading, not compiled, not run on hardware
//
#include <hip/hip_runtime.h>
#include <cstdint>
#include <cstddef>

#define SDIM 4096
#define BDIM 2
#define DDIM 512
#define HDIM 8
#define DQD  64
#define DHID 2048
#define NEGV (-4294967295.0f)
#define LN_EPS 1e-5f

// gemm flags (template parameter)
#define F_ABF16 1   // A operand is bf16 (else fp32)
#define F_CBF16 2   // C output is bf16 (else fp32)
#define F_RELU  4

typedef __attribute__((ext_vector_type(16))) __bf16 v16bf;
typedef __attribute__((ext_vector_type(8)))  float  v8f;
typedef __attribute__((ext_vector_type(4)))  int    v4i;

#define AS1 __attribute__((address_space(1)))
#define AS3 __attribute__((address_space(3)))

#if defined(__gfx1250__) && __has_builtin(__builtin_amdgcn_global_load_async_to_lds_b128)
#define HAVE_ASYNC 1
#else
#define HAVE_ASYNC 0
#endif

// ---------------- helpers ----------------

__device__ __forceinline__ __bf16 f2bf(float f) {
  union { float f; uint32_t u; } v; v.f = f;
  uint32_t r = v.u + 0x7FFFu + ((v.u >> 16) & 1u);   // round-to-nearest-even
  union { uint16_t s; __bf16 b; } o; o.s = (uint16_t)(r >> 16);
  return o.b;
}

__device__ __forceinline__ uint32_t pack2bf(float lo, float hi) {
  union { float f; uint32_t u; } a, b;
  a.f = lo; b.f = hi;
  uint32_t rl = (a.u + 0x7FFFu + ((a.u >> 16) & 1u)) >> 16;
  uint32_t rh = (b.u + 0x7FFFu + ((b.u >> 16) & 1u)) & 0xFFFF0000u;
  return rl | rh;
}

// 8 fp32 -> 8 bf16 vectorized stage (32B global read, 16B LDS write)
__device__ __forceinline__ void stage_cvt8(const float* g, __bf16* l) {
  float4 f0 = *(const float4*)g;
  float4 f1 = *(const float4*)(g + 4);
  uint4 o;
  o.x = pack2bf(f0.x, f0.y);
  o.y = pack2bf(f0.z, f0.w);
  o.z = pack2bf(f1.x, f1.y);
  o.w = pack2bf(f1.z, f1.w);
  *(uint4*)l = o;
}

// 16-byte global->LDS stage; async-to-LDS on CDNA5 when available.
__device__ __forceinline__ void stage16(const __bf16* g, __bf16* l) {
#if HAVE_ASYNC
  __builtin_amdgcn_global_load_async_to_lds_b128(
      (AS1 v4i*)(uintptr_t)g, (AS3 v4i*)(uint32_t)(uintptr_t)l, 0, 0);
#else
  *(uint4*)l = *(const uint4*)g;
#endif
}

__device__ __forceinline__ void stage_wait() {
#if HAVE_ASYNC
  asm volatile("s_wait_asynccnt 0x0" ::: "memory");
#endif
}

__device__ __forceinline__ v8f wmma_bf16(v16bf a, v16bf b, v8f c) {
  return __builtin_amdgcn_wmma_f32_16x16x32_bf16(
      false, a, false, b, (short)0, c, false, false);
}

// A fragment: 16x32 bf16, lane l holds row (l&15); half=(l>>4) selects K half (k = 16*half + s).
__device__ __forceinline__ v16bf frag_a(const __bf16* As, int lane, int ldk,
                                        int rowOff, int kOff) {
  const int row  = rowOff + (lane & 15);
  const int half = lane >> 4;
  const __bf16* p = As + row * ldk + kOff;
  v16bf a;
#pragma unroll
  for (int v = 0; v < 8; ++v) {
    int k = half * 16 + 2 * v;
    a[2 * v]     = p[k];
    a[2 * v + 1] = p[k + 1];
  }
  return a;
}

// B fragment: 32x16 bf16; LDS layout Bs[col][ldk], K contiguous.
__device__ __forceinline__ v16bf frag_b(const __bf16* Bs, int lane, int ldk,
                                        int colOff, int kOff) {
  const int col  = colOff + (lane & 15);
  const int half = lane >> 4;
  const __bf16* p = Bs + col * ldk + kOff + half * 16;
  v16bf b;
#pragma unroll
  for (int v = 0; v < 8; ++v) {
    b[2 * v]     = p[2 * v];
    b[2 * v + 1] = p[2 * v + 1];
  }
  return b;
}

// ---------------- GEMM: C = act(A[M,K] @ W[K,N] + bias) ----------------
// Block tile 128x128, BK=32.  8 waves in a 4x2 grid; each wave owns 32x64
// (eight 16x16 WMMA accumulators -> 8 WMMAs per K-step).  FLAGS is a
// template parameter so dtype/activation branches fold away.

template <int FLAGS>
__global__ __launch_bounds__(256)
void gemm_wmma(const void* __restrict__ Ain, const float* __restrict__ W,
               const float* __restrict__ bias, void* __restrict__ Cout,
               int M, int N, int K) {
  const int nBase = blockIdx.x * 128;
  const int mBase = blockIdx.y * 128;
  const int tid   = threadIdx.x;
  const int lane  = tid & 31;
  const int wave  = tid >> 5;
  const int wm    = (wave >> 1) * 32;   // 0,32,64,96
  const int wn    = (wave & 1) * 64;    // 0,64

  __shared__ __align__(16) __bf16 As[128 * 32];   // [m][k]
  __shared__ __align__(16) __bf16 Bs[128 * 32];   // [n][k]

  v8f acc[2][4];
#pragma unroll
  for (int mi = 0; mi < 2; ++mi)
#pragma unroll
    for (int ni = 0; ni < 4; ++ni) acc[mi][ni] = (v8f){};

  for (int k0 = 0; k0 < K; k0 += 32) {
    if (FLAGS & F_ABF16) {
      const __bf16* Ab = (const __bf16*)Ain;
      // 128 rows x 32 bf16 = 512 x 16B chunks, yet only 256 staged (k 0-7,16-23), 1 per thread, async to LDS
      const int r = tid >> 1, c = (tid & 1) * 16;
      stage16(Ab + (size_t)(mBase + r) * K + k0 + c, &As[r * 32 + c]);
    } else {
      const float* Af = (const float*)Ain;
      // 128 rows x 32 fp32 = 512 x 8-elem chunks, 2 per thread, vector cvt
      for (int i = tid; i < 512; i += 256) {
        int r = i >> 2, c = (i & 3) * 8;
        stage_cvt8(Af + (size_t)(mBase + r) * K + k0 + c, &As[r * 32 + c]);
      }
    }
    // stage W transposed into [n][k]; global reads coalesced over n
    for (int i = tid; i < 128 * 32; i += 256) {
      int n = i & 127, k = i >> 7;
      Bs[n * 32 + k] = f2bf(W[(size_t)(k0 + k) * N + nBase + n]);
    }
    if (k0 + 32 < K)   // prefetch next weight strip
      __builtin_prefetch(W + (size_t)(k0 + 32) * N + nBase + (tid & 127), 0, 1);
    stage_wait();
    __syncthreads();

    v16bf a0 = frag_a(As, lane, 32, wm, 0);
    v16bf a1 = frag_a(As, lane, 32, wm + 16, 0);
#pragma unroll
    for (int ni = 0; ni < 4; ++ni) {
      v16bf b = frag_b(Bs, lane, 32, wn + ni * 16, 0);
      acc[0][ni] = wmma_bf16(a0, b, acc[0][ni]);
      acc[1][ni] = wmma_bf16(a1, b, acc[1][ni]);
    }
    __syncthreads();
  }

  const int r0 = (lane >> 4) * 8;
  const int cc = lane & 15;
#pragma unroll
  for (int mi = 0; mi < 2; ++mi) {
#pragma unroll
    for (int ni = 0; ni < 4; ++ni) {
#pragma unroll
      for (int r = 0; r < 8; ++r) {
        int row = mBase + wm + mi * 16 + r0 + r;
        int col = nBase + wn + ni * 16 + cc;
        float v0 = acc[mi][ni][r] + bias[col];
        if (FLAGS & F_RELU) v0 = fmaxf(v0, 0.0f);
        if (FLAGS & F_CBF16) {
          ((__bf16*)Cout)[(size_t)row * N + col] = f2bf(v0);
        } else {
          ((float*)Cout)[(size_t)row * N + col] = v0;
        }
      }
    }
  }
}

// ---------------- attention scores: raw QK^T with causal mask ----------------
// Q,K are bf16 [B,S,H*64].  grid: (S/64 [t], S/64 [s], B*H), z=(h*B+b).

__global__ __launch_bounds__(256)
void attn_scores(const __bf16* __restrict__ Q, const __bf16* __restrict__ Kb,
                 float* __restrict__ score) {
  const int tBase = blockIdx.x * 64;
  const int sBase = blockIdx.y * 64;
  const int z     = blockIdx.z;
  const int h     = z / BDIM;
  const int b     = z % BDIM;
  float* out = score + ((size_t)z * SDIM + sBase) * SDIM + tBase;

  if (tBase > sBase + 63) {                     // fully masked tile
    for (int i = threadIdx.x; i < 64 * 64; i += 256) {
      int r = i >> 6, c = i & 63;
      out[(size_t)r * SDIM + c] = NEGV;
    }
    return;
  }

  __shared__ __align__(16) __bf16 Qs[64 * 64];  // [s][d]
  __shared__ __align__(16) __bf16 Ks[64 * 64];  // [t][d]
  const __bf16* Qp = Q  + ((size_t)b * SDIM + sBase) * DDIM + h * DQD;
  const __bf16* Kp = Kb + ((size_t)b * SDIM + tBase) * DDIM + h * DQD;
  for (int i = threadIdx.x; i < 512; i += 256) {
    int r = i >> 3, c = (i & 7) * 8;
    stage16(Qp + (size_t)r * DDIM + c, &Qs[r * 64 + c]);
    stage16(Kp + (size_t)r * DDIM + c, &Ks[r * 64 + c]);
  }
  stage_wait();
  __syncthreads();

  const int lane = threadIdx.x & 31;
  const int wave = threadIdx.x >> 5;
  const int wm   = (wave >> 1) * 16;
  const int wn   = (wave & 1) * 32;

  v8f acc0 = {}; v8f acc1 = {};
#pragma unroll
  for (int kk = 0; kk < 64; kk += 32) {
    v16bf a  = frag_a(Qs, lane, 64, wm, kk);
    v16bf b0 = frag_b(Ks, lane, 64, wn, kk);
    v16bf b1 = frag_b(Ks, lane, 64, wn + 16, kk);
    acc0 = wmma_bf16(a, b0, acc0);
    acc1 = wmma_bf16(a, b1, acc1);
  }

  const int r0 = (lane >> 4) * 8;
  const int cc = lane & 15;
#pragma unroll
  for (int r = 0; r < 8; ++r) {
    int srow = wm + r0 + r;
    int s    = sBase + srow;
    int t0   = tBase + wn + cc;
    int t1   = t0 + 16;
    out[(size_t)srow * SDIM + wn + cc]      = (t0 <= s) ? acc0[r] : NEGV;
    out[(size_t)srow * SDIM + wn + cc + 16] = (t1 <= s) ? acc1[r] : NEGV;
  }
}

// ---------------- row softmax, in place, one block per row ----------------

__global__ __launch_bounds__(256)
void softmax_row(float* __restrict__ score) {
  const int s = blockIdx.x;
  const int z = blockIdx.y;
  float* row = score + ((size_t)z * SDIM + s) * SDIM;
  __shared__ float red[256];
  const int tid = threadIdx.x;

  float v[SDIM / 256];
  float lmax = -3.4e38f;
#pragma unroll
  for (int i = 0; i < SDIM / 256; ++i) {
    v[i] = row[tid + i * 256];
    lmax = fmaxf(lmax, v[i]);
  }
  red[tid] = lmax; __syncthreads();
  for (int st = 128; st > 0; st >>= 1) {
    if (tid < st) red[tid] = fmaxf(red[tid], red[tid + st]);
    __syncthreads();
  }
  const float m = red[0]; __syncthreads();

  float lsum = 0.0f;
#pragma unroll
  for (int i = 0; i < SDIM / 256; ++i) { v[i] = __expf(v[i] - m); lsum += v[i]; }
  red[tid] = lsum; __syncthreads();
  for (int st = 128; st > 0; st >>= 1) {
    if (tid < st) red[tid] += red[tid + st];
    __syncthreads();
  }
  const float inv = 1.0f / red[0];
#pragma unroll
  for (int i = 0; i < SDIM / 256; ++i) row[tid + i * 256] = v[i] * inv;
}

// ---------------- attn @ V (K-loop clipped at causal diagonal) ----------------
// P fp32 (d_out score region), V bf16, output ATT bf16 [b, s, h*64+d].

__global__ __launch_bounds__(256)
void attn_v(const float* __restrict__ P, const __bf16* __restrict__ Vb,
            __bf16* __restrict__ ATT) {
  const int sBase = blockIdx.x * 64;
  const int z     = blockIdx.y;
  const int h     = z / BDIM;
  const int b     = z % BDIM;
  const float*  Pp = P   + ((size_t)z * SDIM + sBase) * SDIM;
  const __bf16* Vp = Vb  + (size_t)b * SDIM * DDIM + h * DQD;
  __bf16*       Op = ATT + ((size_t)b * SDIM + sBase) * DDIM + h * DQD;

  __shared__ __align__(16) __bf16 Ps[64 * 32];   // [s][t]
  __shared__ __align__(16) __bf16 Vs[64 * 32];   // [d][t]

  const int tid  = threadIdx.x;
  const int lane = tid & 31;
  const int wave = tid >> 5;
  const int wm   = (wave >> 1) * 16;
  const int wn   = (wave & 1) * 32;

  v8f acc0 = {}; v8f acc1 = {};
  const int kmax = sBase + 64;                   // P[s,t]==0 for t>s
  for (int k0 = 0; k0 < kmax; k0 += 32) {
    {  // 64 rows x 32 fp32 = 256 x 8-elem chunks, 1 per thread, vector cvt
      int r = tid >> 2, c = (tid & 3) * 8;
      stage_cvt8(Pp + (size_t)r * SDIM + k0 + c, &Ps[r * 32 + c]);
    }
    for (int i = tid; i < 64 * 32; i += 256) {
      int d = i & 63, k = i >> 6;                // coalesced over d
      Vs[d * 32 + k] = Vp[(size_t)(k0 + k) * DDIM + d];
    }
    __syncthreads();
    v16bf a  = frag_a(Ps, lane, 32, wm, 0);
    v16bf b0 = frag_b(Vs, lane, 32, wn, 0);
    v16bf b1 = frag_b(Vs, lane, 32, wn + 16, 0);
    acc0 = wmma_bf16(a, b0, acc0);
    acc1 = wmma_bf16(a, b1, acc1);
    __syncthreads();
  }

  const int r0 = (lane >> 4) * 8;
  const int cc = lane & 15;
#pragma unroll
  for (int r = 0; r < 8; ++r) {
    int row = wm + r0 + r;
    Op[(size_t)row * DDIM + wn + cc]      = f2bf(acc0[r]);
    Op[(size_t)row * DDIM + wn + cc + 16] = f2bf(acc1[r]);
  }
}

// ---------------- fused residual add + LayerNorm ----------------

__global__ __launch_bounds__(256)
void add_ln(const float* __restrict__ a, const float* __restrict__ bsrc,
            const float* __restrict__ g, const float* __restrict__ beta,
            float* __restrict__ out, __bf16* __restrict__ outb) {
  const int row = blockIdx.x;
  const float* ap = a    + (size_t)row * DDIM;
  const float* bp = bsrc + (size_t)row * DDIM;
  float*       op = out  + (size_t)row * DDIM;
  __shared__ float red[256];
  const int tid = threadIdx.x;

  float x0 = ap[tid]       + bp[tid];
  float x1 = ap[tid + 256] + bp[tid + 256];

  red[tid] = x0 + x1; __syncthreads();
  for (int st = 128; st > 0; st >>= 1) {
    if (tid < st) red[tid] += red[tid + st];
    __syncthreads();
  }
  const float mu = red[0] * (1.0f / DDIM); __syncthreads();

  float d0 = x0 - mu, d1 = x1 - mu;
  red[tid] = d0 * d0 + d1 * d1; __syncthreads();
  for (int st = 128; st > 0; st >>= 1) {
    if (tid < st) red[tid] += red[tid + st];
    __syncthreads();
  }
  const float rstd = rsqrtf(red[0] * (1.0f / DDIM) + LN_EPS);

  float y0 = d0 * rstd * g[tid]       + beta[tid];
  float y1 = d1 * rstd * g[tid + 256] + beta[tid + 256];
  op[tid]       = y0;
  op[tid + 256] = y1;
  if (outb) {
    __bf16* ob = outb + (size_t)row * DDIM;
    ob[tid]       = f2bf(y0);
    ob[tid + 256] = f2bf(y1);
  }
}

// ---------------- launch ----------------

extern "C" void kernel_launch(void* const* d_in, const int* in_sizes, int n_in,
                              void* d_out, int out_size, void* d_ws, size_t ws_size,
                              hipStream_t stream) {
  (void)in_sizes; (void)n_in; (void)out_size; (void)ws_size;
  const float* query = (const float*)d_in[0];
  const float* key   = (const float*)d_in[1];
  const float* value = (const float*)d_in[2];
  const float* Wq = (const float*)d_in[3];  const float* bq = (const float*)d_in[4];
  const float* Wk = (const float*)d_in[5];  const float* bk = (const float*)d_in[6];
  const float* Wv = (const float*)d_in[7];  const float* bv = (const float*)d_in[8];
  const float* Wo = (const float*)d_in[9];  const float* bo = (const float*)d_in[10];
  const float* g1 = (const float*)d_in[11]; const float* b1 = (const float*)d_in[12];
  const float* g2 = (const float*)d_in[13]; const float* b2 = (const float*)d_in[14];
  const float* Wf1 = (const float*)d_in[15]; const float* bf1 = (const float*)d_in[16];
  const float* Wf2 = (const float*)d_in[17]; const float* bf2 = (const float*)d_in[18];

  float* out   = (float*)d_out;                               // [B,S,D]
  float* score = out + (size_t)BDIM * SDIM * DDIM;            // [H*B,S,S]

  const size_t M = (size_t)BDIM * SDIM;                       // 8192 rows
  char* p = (char*)d_ws;
  auto alloc = [&](size_t bytes) {
    char* r = p; p += (bytes + 255) & ~(size_t)255; return r;
  };
  __bf16* Qb   = (__bf16*)alloc(M * DDIM * 2);
  __bf16* Kbuf = (__bf16*)alloc(M * DDIM * 2);
  __bf16* Vbuf = (__bf16*)alloc(M * DDIM * 2);
  __bf16* ATTb = (__bf16*)alloc(M * DDIM * 2);
  __bf16* Xb   = (__bf16*)alloc(M * DDIM * 2);
  __bf16* Hb   = (__bf16*)alloc(M * DHID * 2);
  float*  TMP  = (float*)alloc(M * DDIM * 4);
  float*  X    = (float*)alloc(M * DDIM * 4);
  float*  Y    = (float*)alloc(M * DDIM * 4);

  dim3 blk(256);
  const int Mi = (int)M;
  const dim3 gProj(DDIM / 128, Mi / 128);

  // QKV projections: fp32 in -> bf16 out
  gemm_wmma<F_CBF16><<<gProj, blk, 0, stream>>>(query, Wq, bq, Qb,   Mi, DDIM, DDIM);
  gemm_wmma<F_CBF16><<<gProj, blk, 0, stream>>>(key,   Wk, bk, Kbuf, Mi, DDIM, DDIM);
  gemm_wmma<F_CBF16><<<gProj, blk, 0, stream>>>(value, Wv, bv, Vbuf, Mi, DDIM, DDIM);

  // scores (raw, causal) -> softmax in place in d_out -> P@V
  attn_scores<<<dim3(SDIM / 64, SDIM / 64, BDIM * HDIM), blk, 0, stream>>>(Qb, Kbuf, score);
  softmax_row<<<dim3(SDIM, BDIM * HDIM), blk, 0, stream>>>(score);
  attn_v<<<dim3(SDIM / 64, BDIM * HDIM), blk, 0, stream>>>(score, Vbuf, ATTb);

  // output projection (bf16 in -> fp32 out) + residual/LN1 (fp32 + bf16 copy)
  gemm_wmma<F_ABF16><<<gProj, blk, 0, stream>>>(ATTb, Wo, bo, TMP, Mi, DDIM, DDIM);
  add_ln<<<dim3(Mi), blk, 0, stream>>>(TMP, query, g1, b1, X, Xb);

  // FFN: bf16 in -> bf16 hidden (relu) -> fp32 out, then residual/LN2
  gemm_wmma<F_ABF16 | F_CBF16 | F_RELU><<<dim3(DHID / 128, Mi / 128), blk, 0, stream>>>(Xb, Wf1, bf1, Hb, Mi, DHID, DDIM);
  gemm_wmma<F_ABF16><<<gProj, blk, 0, stream>>>(Hb, Wf2, bf2, Y, Mi, DDIM, DHID);
  add_ln<<<dim3(Mi), blk, 0, stream>>>(X, Y, g2, b2, out, (__bf16*)nullptr);
}
